// FinancialGNN_26620207300736
// MI455X (gfx1250) — matrix operand order repaired and checked by reading, not compiled, not run on hardware
//
#include <hip/hip_runtime.h>
#include <hip/hip_bf16.h>
#include <math.h>

#define NCC 100000
#define NEE 150000
#define EDG 600000
#define HID 128
#define HEADS_N 4
#define HDIM 32
#define DOUT 64

typedef __attribute__((ext_vector_type(16))) _Float16 v16h;
typedef __attribute__((ext_vector_type(8)))  float    v8f;

// ---------------- monotone float<->uint encoding for atomicMax on floats ----
static __device__ __forceinline__ unsigned f2ord(float f) {
  unsigned u = __float_as_uint(f);
  return (u & 0x80000000u) ? ~u : (u | 0x80000000u);
}
static __device__ __forceinline__ float ord2f(unsigned u) {
  return __uint_as_float((u & 0x80000000u) ? (u & 0x7fffffffu) : ~u);
}

// ---------------- WMMA GEMM: C[M,N] (+)= act( A[M,128] @ W[128,N] + bias ) --
// flags bit0: accumulate into existing C; bit1: relu epilogue.
// 256 threads = 8 waves; each wave computes 16 rows x N cols via
// v_wmma_f32_16x16x32_f16 (4 k-steps). W staged in LDS in B-fragment order.
// All four A fragments are hoisted into registers up front so the 4*NT WMMA
// chain only waits on LDS feeds, not global latency.
template <int N>
__global__ __launch_bounds__(256) void gemm_wmma(
    const float* __restrict__ A, const float* __restrict__ W,
    const float* __restrict__ bias, float* __restrict__ C, int M, int flags) {
  constexpr int NT = N / 16;
  __shared__ _Float16 wl[4 * NT * 32 * 16];
  const int tid = threadIdx.x;

  // Stage W -> LDS, pre-swizzled: element t = ((kt*NT+nt)*32 + lane)*16 + idx
  // B layout (16-bit, 32x16): lane group g holds K = (idx>>3)*16 + g*8 + (idx&7), Ncol = lane&15.
  for (int t = tid; t < 4 * NT * 512; t += 256) {
    int idx  = t & 15;
    int lane = (t >> 4) & 31;
    int tb   = t >> 9;           // kt*NT + nt
    int nt   = tb % NT;
    int kt   = tb / NT;
    int k    = kt * 32 + (idx >> 3) * 16 + (lane >> 4) * 8 + (idx & 7);
    int ncol = nt * 16 + (lane & 15);
    wl[t] = (_Float16)W[(size_t)k * N + ncol];
  }

  const int lane = tid & 31;
  const int wave = tid >> 5;
  const int g    = lane >> 4;
  const int rowb = blockIdx.x * 128 + wave * 16;

  int ar = rowb + (lane & 15);
  if (ar > M - 1) ar = M - 1;          // clamp: keep EXEC all-ones for WMMA
  const float* arow = A + (size_t)ar * HID;

  // Prefetch next block's A stripe into cache (global_prefetch_b8).
  {
    int pr = ar + 128;
    if (pr > M - 1) pr = M - 1;
    __builtin_prefetch(A + (size_t)pr * HID, 0, 1);
  }

  // Hoist all A fragments: 16 independent b128 loads + packed f32->f16 cvt.
  v16h afr[4];
#pragma unroll
  for (int kt = 0; kt < 4; ++kt) {
    // A layout (16-bit, 16x32): idx 0..7 -> K=g*8+idx ; idx 8..15 -> K=16+g*8+(idx-8)
    const float* ap = arow + kt * 32 + g * 8;
    float4 x0 = *(const float4*)(ap);
    float4 x1 = *(const float4*)(ap + 4);
    float4 x2 = *(const float4*)(ap + 16);
    float4 x3 = *(const float4*)(ap + 20);
    v16h a;
    a[0]=(_Float16)x0.x; a[1]=(_Float16)x0.y; a[2]=(_Float16)x0.z; a[3]=(_Float16)x0.w;
    a[4]=(_Float16)x1.x; a[5]=(_Float16)x1.y; a[6]=(_Float16)x1.z; a[7]=(_Float16)x1.w;
    a[8]=(_Float16)x2.x; a[9]=(_Float16)x2.y; a[10]=(_Float16)x2.z; a[11]=(_Float16)x2.w;
    a[12]=(_Float16)x3.x; a[13]=(_Float16)x3.y; a[14]=(_Float16)x3.z; a[15]=(_Float16)x3.w;
    afr[kt] = a;
  }

  __syncthreads();

  v8f acc[NT];
  for (int i = 0; i < NT; ++i) { v8f z = {}; acc[i] = z; }

#pragma unroll
  for (int kt = 0; kt < 4; ++kt) {
#pragma unroll
    for (int nt = 0; nt < NT; ++nt) {
      const v16h b = *(const v16h*)&wl[((kt * NT + nt) * 32 + lane) * 16];
      acc[nt] = __builtin_amdgcn_wmma_f32_16x16x32_f16(
          false, afr[kt], false, b, (short)0, acc[nt], false, false);
    }
  }

  // D layout: v8f element r -> row = rowb + g*8 + r, col = nt*16 + (lane&15)
  const int crow = rowb + g * 8;
  const int cin  = lane & 15;
  for (int nt = 0; nt < NT; ++nt) {
    int col = nt * 16 + cin;
    float bv = bias ? bias[col] : 0.f;
#pragma unroll
    for (int r = 0; r < 8; ++r) {
      int row = crow + r;
      if (row < M) {
        float v = acc[nt][r] + bv;
        if (flags & 1) v += C[(size_t)row * N + col];
        if (flags & 2) v = fmaxf(v, 0.f);
        C[(size_t)row * N + col] = v;
      }
    }
  }
}

// ---------------- small elementwise / edge kernels ---------------------------
__global__ void encode_kernel(const float* __restrict__ X, const float* __restrict__ W,
                              const float* __restrict__ b, float* __restrict__ Y,
                              int Nn, int Kin) {
  long i = (long)blockIdx.x * blockDim.x + threadIdx.x;
  if (i >= (long)Nn * HID) return;
  int n = (int)(i >> 7), c = (int)(i & 127);
  float acc = b[c];
  for (int k = 0; k < Kin; ++k) acc += X[(long)n * Kin + k] * W[k * HID + c];
  Y[i] = fmaxf(acc, 0.f);
}

__global__ void init_bias3(float* __restrict__ Y, const float* __restrict__ b0,
                           const float* __restrict__ b1, const float* __restrict__ b2,
                           long n) {
  long i = (long)blockIdx.x * blockDim.x + threadIdx.x;
  if (i >= n) return;
  int c = (int)(i & 127);
  Y[i] = b0[c] + b1[c] + b2[c];
}

__global__ void fill_f32(float* p, float v, long n) {
  long i = (long)blockIdx.x * blockDim.x + threadIdx.x;
  if (i < n) p[i] = v;
}
__global__ void fill_u32(unsigned* p, unsigned v, long n) {
  long i = (long)blockIdx.x * blockDim.x + threadIdx.x;
  if (i < n) p[i] = v;
}
__global__ void relu_k(float* p, long n) {
  long i = (long)blockIdx.x * blockDim.x + threadIdx.x;
  if (i < n) p[i] = fmaxf(p[i], 0.f);
}

// a_src[n,h] = sum_d Hx[n, h*32+d] * av[h,d]
__global__ void att_coef(const float* __restrict__ Hx, const float* __restrict__ av,
                         float* __restrict__ o, int Nn) {
  long i = (long)blockIdx.x * blockDim.x + threadIdx.x;
  if (i >= (long)Nn * 4) return;
  int n = (int)(i >> 2), h = (int)(i & 3);
  const float* hp = Hx + (long)n * HID + h * HDIM;
  const float* ap = av + h * HDIM;
  float s = 0.f;
#pragma unroll
  for (int d = 0; d < HDIM; ++d) s += hp[d] * ap[d];
  o[i] = s;
}

__global__ void gat_max(const int* __restrict__ ei, int ne,
                        const float* __restrict__ as_, const float* __restrict__ ad_,
                        unsigned* __restrict__ mv) {
  int e = blockIdx.x * blockDim.x + threadIdx.x;
  if (e >= ne) return;
  int s = ei[e], d = ei[ne + e];
#pragma unroll
  for (int h = 0; h < 4; ++h) {
    float v = as_[s * 4 + h] + ad_[d * 4 + h];
    v = v >= 0.f ? v : 0.2f * v;
    atomicMax(&mv[d * 4 + h], f2ord(v));
  }
}

__global__ void gat_den(const int* __restrict__ ei, int ne,
                        const float* __restrict__ as_, const float* __restrict__ ad_,
                        const unsigned* __restrict__ mv, float* __restrict__ den) {
  int e = blockIdx.x * blockDim.x + threadIdx.x;
  if (e >= ne) return;
  int s = ei[e], d = ei[ne + e];
#pragma unroll
  for (int h = 0; h < 4; ++h) {
    float v = as_[s * 4 + h] + ad_[d * 4 + h];
    v = v >= 0.f ? v : 0.2f * v;
    atomicAdd(&den[d * 4 + h], __expf(v - ord2f(mv[d * 4 + h])));
  }
}

// 128 threads per edge: out[d,:] += alpha[h(c)] * hs[s,:]
__global__ void gat_scatter(const int* __restrict__ ei, int ne,
                            const float* __restrict__ as_, const float* __restrict__ ad_,
                            const unsigned* __restrict__ mv, const float* __restrict__ den,
                            const float* __restrict__ hs, float* __restrict__ out) {
  int e = blockIdx.x * 2 + (threadIdx.x >> 7);
  if (e >= ne) return;
  int c = threadIdx.x & 127;
  int h = c >> 5;
  int s = ei[e], d = ei[ne + e];
  float v = as_[s * 4 + h] + ad_[d * 4 + h];
  v = v >= 0.f ? v : 0.2f * v;
  float alpha = __expf(v - ord2f(mv[d * 4 + h])) / den[d * 4 + h];
  atomicAdd(&out[(long)d * HID + c], hs[(long)s * HID + c] * alpha);
}

__global__ void deg_count(const int* __restrict__ ei, int ne, float* __restrict__ deg) {
  int e = blockIdx.x * blockDim.x + threadIdx.x;
  if (e >= ne) return;
  atomicAdd(&deg[ei[ne + e]], 1.f);
}

__global__ void seg_sum(const int* __restrict__ ei, int ne,
                        const float* __restrict__ X, float* __restrict__ out) {
  int e = blockIdx.x * 2 + (threadIdx.x >> 7);
  if (e >= ne) return;
  int c = threadIdx.x & 127;
  int s = ei[e], d = ei[ne + e];
  atomicAdd(&out[(long)d * HID + c], X[(long)s * HID + c]);
}

__global__ void mean_div(float* __restrict__ agg, const float* __restrict__ deg, long n) {
  long i = (long)blockIdx.x * blockDim.x + threadIdx.x;
  if (i >= n) return;
  agg[i] = agg[i] / fmaxf(deg[i >> 7], 1.f);
}

__global__ void norm_rows(const float* __restrict__ X, float* __restrict__ o, int Nn) {
  int n = blockIdx.x * blockDim.x + threadIdx.x;
  if (n >= Nn) return;
  const float* xp = X + (long)n * DOUT;
  float ss = 0.f;
#pragma unroll
  for (int c = 0; c < DOUT; ++c) ss += xp[c] * xp[c];
  float inv = 1.f / fmaxf(sqrtf(ss), 1e-12f);
  float* op = o + (long)n * DOUT;
#pragma unroll
  for (int c = 0; c < DOUT; ++c) op[c] = xp[c] * inv;
}

// ---------------- host orchestration ----------------------------------------
extern "C" void kernel_launch(void* const* d_in, const int* in_sizes, int n_in,
                              void* d_out, int out_size, void* d_ws, size_t ws_size,
                              hipStream_t stream) {
  (void)in_sizes; (void)n_in; (void)out_size; (void)ws_size;
  const float* xc     = (const float*)d_in[0];
  const float* xe     = (const float*)d_in[1];
  const float* Wc     = (const float*)d_in[2];
  const float* bc     = (const float*)d_in[3];
  const float* We     = (const float*)d_in[4];
  const float* be     = (const float*)d_in[5];
  const float* gat_W  = (const float*)d_in[6];
  const float* gat_as = (const float*)d_in[7];
  const float* gat_ad = (const float*)d_in[8];
  const float* gat_b  = (const float*)d_in[9];
  const float* s2_Wl  = (const float*)d_in[10];
  const float* s2_bl  = (const float*)d_in[11];
  const float* s2_Wr  = (const float*)d_in[12];
  const float* s3_Wl  = (const float*)d_in[13];
  const float* s3_bl  = (const float*)d_in[14];
  const float* s3_Wr  = (const float*)d_in[15];
  const int* ei[6] = {(const int*)d_in[16], (const int*)d_in[17], (const int*)d_in[18],
                      (const int*)d_in[19], (const int*)d_in[20], (const int*)d_in[21]};
  float* out = (float*)d_out;

  char* wsp = (char*)d_ws;
  size_t off = 0;
  auto take = [&](size_t elems) -> float* {
    float* p = (float*)(wsp + off);
    off += ((elems * 4 + 255) & ~(size_t)255);
    return p;
  };
  float*    hc   = take((size_t)NCC * HID);
  float*    he   = take((size_t)NEE * HID);
  float*    c1   = take((size_t)NCC * HID);
  float*    e1   = take((size_t)NEE * HID);
  float*    t1   = take((size_t)NEE * HID);   // hs / agg / mean
  float*    t2   = take((size_t)NEE * HID);   // hd
  float*    asrc = take((size_t)NEE * 4);
  float*    adst = take((size_t)NEE * 4);
  unsigned* mval = (unsigned*)take((size_t)NEE * 4);
  float*    den  = take((size_t)NEE * 4);
  float*    deg  = take((size_t)NEE);
  float* c2 = hc; float* e2 = he;   // hc/he dead after layer 1
  float* c3 = c1; float* e3 = e1;   // c1/e1 dead after layer 2

  const int B = 256;
  auto cdiv = [](long a, long b) { return (int)((a + b - 1) / b); };

  // ---- layer 0: feature encoders
  encode_kernel<<<cdiv((long)NCC * HID, B), B, 0, stream>>>(xc, Wc, bc, hc, NCC, 5);
  encode_kernel<<<cdiv((long)NEE * HID, B), B, 0, stream>>>(xe, We, be, he, NEE, 4);

  // ---- layer 1: hetero GAT (accumulators init = sum of per-type biases)
  init_bias3<<<cdiv((long)NCC * HID, B), B, 0, stream>>>(c1, gat_b, gat_b + HID, gat_b + 2 * HID, (long)NCC * HID);
  init_bias3<<<cdiv((long)NEE * HID, B), B, 0, stream>>>(e1, gat_b + 3 * HID, gat_b + 4 * HID, gat_b + 5 * HID, (long)NEE * HID);

  struct L1T { int t; const float* xs; const float* xd; int Ns; int Nd; float* o; const int* e; };
  L1T L1[6] = {
    {0, he, hc, NEE, NCC, c1, ei[0]}, {1, he, hc, NEE, NCC, c1, ei[1]},
    {2, he, hc, NEE, NCC, c1, ei[2]}, {3, he, he, NEE, NEE, e1, ei[3]},
    {4, hc, he, NCC, NEE, e1, ei[4]}, {5, hc, he, NCC, NEE, e1, ei[5]},
  };
  for (int i = 0; i < 6; ++i) {
    const L1T& L = L1[i];
    const float* Wt = gat_W + (size_t)L.t * HID * HID;
    gemm_wmma<128><<<cdiv(L.Ns, 128), 256, 0, stream>>>(L.xs, Wt, nullptr, t1, L.Ns, 0);
    const float* hd = t1;
    if (L.xs != L.xd) {
      gemm_wmma<128><<<cdiv(L.Nd, 128), 256, 0, stream>>>(L.xd, Wt, nullptr, t2, L.Nd, 0);
      hd = t2;
    }
    att_coef<<<cdiv((long)L.Ns * 4, B), B, 0, stream>>>(t1, gat_as + (size_t)L.t * HEADS_N * HDIM, asrc, L.Ns);
    att_coef<<<cdiv((long)L.Nd * 4, B), B, 0, stream>>>(hd, gat_ad + (size_t)L.t * HEADS_N * HDIM, adst, L.Nd);
    fill_u32<<<cdiv((long)L.Nd * 4, B), B, 0, stream>>>(mval, 0u, (long)L.Nd * 4);
    fill_f32<<<cdiv((long)L.Nd * 4, B), B, 0, stream>>>(den, 0.f, (long)L.Nd * 4);
    gat_max<<<cdiv(EDG, B), B, 0, stream>>>(L.e, EDG, asrc, adst, mval);
    gat_den<<<cdiv(EDG, B), B, 0, stream>>>(L.e, EDG, asrc, adst, mval, den);
    gat_scatter<<<(EDG + 1) / 2, 256, 0, stream>>>(L.e, EDG, asrc, adst, mval, den, t1, L.o);
  }
  relu_k<<<cdiv((long)NCC * HID, B), B, 0, stream>>>(c1, (long)NCC * HID);
  relu_k<<<cdiv((long)NEE * HID, B), B, 0, stream>>>(e1, (long)NEE * HID);

  // ---- layer 2: hetero SAGE (H -> H)
  struct S2T { int t; const float* xs; const int* e; };
  {  // company dst (types 0,1,2; src = e1; xd = c1)
    S2T cs[3] = {{0, e1, ei[0]}, {1, e1, ei[1]}, {2, e1, ei[2]}};
    for (int i = 0; i < 3; ++i) {
      fill_f32<<<cdiv((long)NCC, B), B, 0, stream>>>(deg, 0.f, (long)NCC);
      deg_count<<<cdiv(EDG, B), B, 0, stream>>>(cs[i].e, EDG, deg);
      fill_f32<<<cdiv((long)NCC * HID, B), B, 0, stream>>>(t1, 0.f, (long)NCC * HID);
      seg_sum<<<(EDG + 1) / 2, 256, 0, stream>>>(cs[i].e, EDG, cs[i].xs, t1);
      mean_div<<<cdiv((long)NCC * HID, B), B, 0, stream>>>(t1, deg, (long)NCC * HID);
      gemm_wmma<128><<<cdiv(NCC, 128), 256, 0, stream>>>(
          t1, s2_Wl + (size_t)cs[i].t * HID * HID, s2_bl + (size_t)cs[i].t * HID, c2, NCC, i == 0 ? 0 : 1);
    }
    for (int i = 0; i < 3; ++i)
      gemm_wmma<128><<<cdiv(NCC, 128), 256, 0, stream>>>(
          c1, s2_Wr + (size_t)i * HID * HID, nullptr, c2, NCC, 1 | (i == 2 ? 2 : 0));
  }
  {  // entity dst (types 3,4,5; src = e1,c1,c1; xd = e1)
    S2T es[3] = {{3, e1, ei[3]}, {4, c1, ei[4]}, {5, c1, ei[5]}};
    for (int i = 0; i < 3; ++i) {
      fill_f32<<<cdiv((long)NEE, B), B, 0, stream>>>(deg, 0.f, (long)NEE);
      deg_count<<<cdiv(EDG, B), B, 0, stream>>>(es[i].e, EDG, deg);
      fill_f32<<<cdiv((long)NEE * HID, B), B, 0, stream>>>(t1, 0.f, (long)NEE * HID);
      seg_sum<<<(EDG + 1) / 2, 256, 0, stream>>>(es[i].e, EDG, es[i].xs, t1);
      mean_div<<<cdiv((long)NEE * HID, B), B, 0, stream>>>(t1, deg, (long)NEE * HID);
      gemm_wmma<128><<<cdiv(NEE, 128), 256, 0, stream>>>(
          t1, s2_Wl + (size_t)es[i].t * HID * HID, s2_bl + (size_t)es[i].t * HID, e2, NEE, i == 0 ? 0 : 1);
    }
    for (int i = 0; i < 3; ++i)
      gemm_wmma<128><<<cdiv(NEE, 128), 256, 0, stream>>>(
          e1, s2_Wr + (size_t)(3 + i) * HID * HID, nullptr, e2, NEE, 1 | (i == 2 ? 2 : 0));
  }

  // ---- layer 3: hetero SAGE (H -> D), types {0:emp, 1:board, 2:codir, 3:revemp}
  {  // company dst: types 0,1; src = e2; xd = c2
    for (int i = 0; i < 2; ++i) {
      fill_f32<<<cdiv((long)NCC, B), B, 0, stream>>>(deg, 0.f, (long)NCC);
      deg_count<<<cdiv(EDG, B), B, 0, stream>>>(ei[i], EDG, deg);
      fill_f32<<<cdiv((long)NCC * HID, B), B, 0, stream>>>(t1, 0.f, (long)NCC * HID);
      seg_sum<<<(EDG + 1) / 2, 256, 0, stream>>>(ei[i], EDG, e2, t1);
      mean_div<<<cdiv((long)NCC * HID, B), B, 0, stream>>>(t1, deg, (long)NCC * HID);
      gemm_wmma<64><<<cdiv(NCC, 128), 256, 0, stream>>>(
          t1, s3_Wl + (size_t)i * HID * DOUT, s3_bl + (size_t)i * DOUT, c3, NCC, i == 0 ? 0 : 1);
    }
    for (int i = 0; i < 2; ++i)
      gemm_wmma<64><<<cdiv(NCC, 128), 256, 0, stream>>>(
          c2, s3_Wr + (size_t)i * HID * DOUT, nullptr, c3, NCC, 1);
  }
  {  // entity dst: type 2 (src e2, ei_codir), type 3 (src c2, ei_revemp); xd = e2
    const float* srcs[2] = {e2, c2};
    const int* eis[2] = {ei[3], ei[4]};
    for (int i = 0; i < 2; ++i) {
      fill_f32<<<cdiv((long)NEE, B), B, 0, stream>>>(deg, 0.f, (long)NEE);
      deg_count<<<cdiv(EDG, B), B, 0, stream>>>(eis[i], EDG, deg);
      fill_f32<<<cdiv((long)NEE * HID, B), B, 0, stream>>>(t1, 0.f, (long)NEE * HID);
      seg_sum<<<(EDG + 1) / 2, 256, 0, stream>>>(eis[i], EDG, srcs[i], t1);
      mean_div<<<cdiv((long)NEE * HID, B), B, 0, stream>>>(t1, deg, (long)NEE * HID);
      gemm_wmma<64><<<cdiv(NEE, 128), 256, 0, stream>>>(
          t1, s3_Wl + (size_t)(2 + i) * HID * DOUT, s3_bl + (size_t)(2 + i) * DOUT, e3, NEE, i == 0 ? 0 : 1);
    }
    for (int i = 0; i < 2; ++i)
      gemm_wmma<64><<<cdiv(NEE, 128), 256, 0, stream>>>(
          e2, s3_Wr + (size_t)(2 + i) * HID * DOUT, nullptr, e3, NEE, 1);
  }

  // ---- L2-normalize and write outputs
  norm_rows<<<cdiv((long)NCC, B), B, 0, stream>>>(c3, out, NCC);
  norm_rows<<<cdiv((long)NEE, B), B, 0, stream>>>(e3, out + (size_t)NCC * DOUT, NEE);
}
